// Block_Order_Aware_Filtering_1_to_1_90460601188529
// MI455X (gfx1250) — matrix operand order repaired and checked by reading, not compiled, not run on hardware
//
#include <hip/hip_runtime.h>
#include <stdint.h>

typedef unsigned short u16;
typedef __attribute__((ext_vector_type(16))) __bf16 bf16x16;
typedef __attribute__((ext_vector_type(8)))  float   f32x8;

union AB { bf16x16 v; u16 u[16]; unsigned w[8]; uint4 q[2]; };
union CD { f32x8 v; float f[8]; };

constexpr int B_  = 32;
constexpr int C_  = 128;
constexpr int N_  = 4096;
constexpr int M_  = 512;
constexpr int L_  = 6;
constexpr int NT  = 64;       // pool N tile
constexpr int NT2 = 64;       // unpool N tile
constexpr int CP  = C_ + 8;   // padded K-stride (u16) for K=C tiles
constexpr int NP  = NT + 8;   // padded K-stride (u16) for K=N tiles

__device__ __forceinline__ u16 f2b(float f) {
  unsigned x = __float_as_uint(f);
  x += 0x7FFFu + ((x >> 16) & 1u);          // round-to-nearest-even
  return (u16)(x >> 16);
}

__device__ __forceinline__ f32x8 wmma_bf16(bf16x16 a, bf16x16 b, f32x8 c) {
  return __builtin_amdgcn_wmma_f32_16x16x32_bf16(false, a, false, b, (short)0, c, false, false);
}

// A-frag: two 16B runs at +h and +16+h (h = 8 for upper lane half)
__device__ __forceinline__ void load_afrag_g(AB& a, const u16* base, int h8) {
  a.q[0] = *(const uint4*)(base + h8);
  a.q[1] = *(const uint4*)(base + 16 + h8);
}
// B-frag: two 16B runs at +h and +16+h (h = 8 for upper lane half)
__device__ __forceinline__ void load_bfrag_g(AB& b, const u16* base, int k16) {
  b.q[0] = *(const uint4*)(base + (k16 >> 1));
  b.q[1] = *(const uint4*)(base + 16 + (k16 >> 1));
}

// CDNA5 async global->LDS copy (ASYNCcnt-tracked, no VGPR data round-trip)
__device__ __forceinline__ void async_ld_b128(unsigned lds_addr, const u16* g) {
  asm volatile("global_load_async_to_lds_b128 %0, %1, off"
               :: "v"(lds_addr), "v"(g) : "memory");
}
#define WAIT_ASYNC(n) asm volatile("s_wait_asynccnt %0" :: "i"(n) : "memory")

// ---------------- weight conversion + BN-stats zeroing ----------------
__global__ __launch_bounds__(256) void prep_kernel(
    const float* __restrict__ Wp, const float* __restrict__ Wf,
    const float* __restrict__ Wu,
    u16* __restrict__ Wpb, u16* __restrict__ Wfb, u16* __restrict__ Wub,
    float* __restrict__ stats)
{
  long long i = (long long)blockIdx.x * blockDim.x + threadIdx.x;
  const long long nWp = (long long)M_ * C_;
  const long long nWf = (long long)L_ * M_ * M_;
  const long long nWu = (long long)M_ * C_;
  const long long nSt = (long long)L_ * 2 * M_;
  if (i < nWp) { Wpb[i] = f2b(Wp[i]); return; }
  i -= nWp;
  if (i < nWf) { Wfb[i] = f2b(Wf[i]); return; }
  i -= nWf;
  if (i < nWu) { Wub[i] = f2b(Wu[i]); return; }
  i -= nWu;
  if (i < nSt) { stats[i] = 0.f; }
}

// ---------------- xs -> bf16 [B,C,N] and transposed [B,N,C] ----------------
__global__ __launch_bounds__(256) void convert_kernel(
    const float* __restrict__ x, u16* __restrict__ xsb, u16* __restrict__ xsbT)
{
  __shared__ float tile[32][33];
  const int b = blockIdx.x, c0 = blockIdx.y * 32, n0 = blockIdx.z * 32;
  const int tx = threadIdx.x, ty = threadIdx.y;
  #pragma unroll
  for (int i = ty; i < 32; i += 8) {
    const float v = x[((size_t)b * C_ + c0 + i) * N_ + n0 + tx];
    tile[i][tx] = v;
    xsb[((size_t)b * C_ + c0 + i) * N_ + n0 + tx] = f2b(v);
  }
  __syncthreads();
  #pragma unroll
  for (int i = ty; i < 32; i += 8)
    xsbT[((size_t)b * N_ + n0 + i) * C_ + c0 + tx] = f2b(tile[tx][i]);
}

// ------- fused pool: logits -> exp -> hT = xs @ E^T, row-normalize -------
// outputs hT in [B][C][M] layout (f32 residual + bf16 operand copy)
__global__ __launch_bounds__(128) void pool_kernel(
    const u16* __restrict__ xsb, const u16* __restrict__ xsbT,
    const u16* __restrict__ Wpb, float* __restrict__ hTf, u16* __restrict__ hTb)
{
  const int b = blockIdx.x;
  const int mblk = blockIdx.y;
  const int tid = threadIdx.x, wave = tid >> 5, lane = tid & 31;
  const int half8 = (lane >= 16) ? 8 : 0, col = lane & 15;
  const int h8 = half8, k16 = (lane >= 16) ? 16 : 0;
  const int mbase = mblk * 64 + wave * 16;
  constexpr int NIT = N_ / NT;

  __shared__ u16 tile_nc[2][NT][CP];  // [n][c] : K=c contiguous (logits B)
  __shared__ u16 tile_cn[2][C_][NP];  // [c][n] : K=n contiguous (hT A)
  __shared__ u16 et[4][16][NP];       // per-wave E [m][n] : K=n contiguous
  __shared__ float Zs[4][16];

  const u16* xrow  = xsb  + (size_t)b * C_ * N_;
  const u16* xTrow = xsbT + (size_t)b * N_ * C_;

  // per-thread async-copy addressing (8 + 8 b128 per thread per tile)
  const int rr = tid >> 1, hh = (tid & 1) * 64;
  unsigned lds_nc[2], lds_cn[2];
  lds_nc[0] = (unsigned)(size_t)&tile_nc[0][rr][hh];
  lds_nc[1] = (unsigned)(size_t)&tile_nc[1][rr][hh];
  lds_cn[0] = (unsigned)(size_t)&tile_cn[0][tid][0];
  lds_cn[1] = (unsigned)(size_t)&tile_cn[1][tid][0];

  auto issue_tile = [&](int buf, int n0) {
    const u16* gT = xTrow + (size_t)(n0 + rr) * C_ + hh;
    const u16* g  = xrow + (size_t)tid * N_ + n0;
    #pragma unroll
    for (int i = 0; i < 8; ++i) async_ld_b128(lds_nc[buf] + i * 16, gT + i * 8);
    #pragma unroll
    for (int i = 0; i < 8; ++i) async_ld_b128(lds_cn[buf] + i * 16, g + i * 8);
  };

  // hoist W_pool A-fragments (loop-invariant over N)
  AB aw[4];
  #pragma unroll
  for (int kc = 0; kc < 4; ++kc)
    load_afrag_g(aw[kc], Wpb + (size_t)(mbase + col) * C_ + kc * 32, h8);

  CD hacc[8];
  #pragma unroll
  for (int t = 0; t < 8; ++t)
    #pragma unroll
    for (int r = 0; r < 8; ++r) hacc[t].f[r] = 0.f;
  float Z[8] = {0.f, 0.f, 0.f, 0.f, 0.f, 0.f, 0.f, 0.f};

  issue_tile(0, 0);

  for (int it = 0; it < NIT; ++it) {
    const int cur = it & 1;
    __syncthreads();                      // prev readers of buf cur^1 are done
    if (it + 1 < NIT) {
      issue_tile(cur ^ 1, (it + 1) * NT); // prefetch next tile (overlaps compute)
      WAIT_ASYNC(16);                     // in-order: current tile's batch landed
    } else {
      WAIT_ASYNC(0);
    }
    __syncthreads();                      // all waves' async writes visible

    const u16 (*tnc)[CP] = tile_nc[cur];
    const u16 (*tcn)[NP] = tile_cn[cur];

    // logits tile: 16(m) x 64(n), K = C = 128
    CD lt[4];
    #pragma unroll
    for (int t = 0; t < 4; ++t)
      #pragma unroll
      for (int r = 0; r < 8; ++r) lt[t].f[r] = 0.f;

    #pragma unroll
    for (int kc = 0; kc < 4; ++kc)
      #pragma unroll
      for (int t = 0; t < 4; ++t) {
        AB bb;
        load_bfrag_g(bb, &tnc[t * 16 + col][kc * 32], k16);
        lt[t].v = wmma_bf16(aw[kc].v, bb.v, lt[t].v);
      }

    // E = exp(logit); row-sum partials; stash bf16 E in LDS [m][n]
    #pragma unroll
    for (int t = 0; t < 4; ++t)
      #pragma unroll
      for (int r = 0; r < 8; ++r) {
        const float e = __expf(lt[t].f[r]);
        Z[r] += e;
        et[wave][r + half8][t * 16 + col] = f2b(e);
      }

    // hacc(c x m) += xs(16c x NTn) @ E^T(NTn x 16m)
    #pragma unroll
    for (int kn = 0; kn < 2; ++kn) {
      AB be;   // B-frag: col = m (this wave's 16 rows), K = n
      load_bfrag_g(be, &et[wave][col][kn * 32], k16);
      #pragma unroll
      for (int t = 0; t < 8; ++t) {
        AB ax;   // A-frag: row = c, K = n
        load_afrag_g(ax, &tcn[t * 16 + col][kn * 32], h8);
        hacc[t].v = wmma_bf16(ax.v, be.v, hacc[t].v);
      }
    }
  }

  // reduce Z across the 16 lanes sharing each m-row, broadcast per column
  #pragma unroll
  for (int r = 0; r < 8; ++r) {
    float z = Z[r];
    z += __shfl_xor(z, 1);
    z += __shfl_xor(z, 2);
    z += __shfl_xor(z, 4);
    z += __shfl_xor(z, 8);
    Z[r] = z;
  }
  if (col == 0) {
    #pragma unroll
    for (int r = 0; r < 8; ++r) Zs[wave][r + half8] = Z[r];
  }
  const float zinv = 1.f / Zs[wave][col];

  const int m = mbase + col;
  #pragma unroll
  for (int t = 0; t < 8; ++t)
    #pragma unroll
    for (int r = 0; r < 8; ++r) {
      const int c = t * 16 + r + half8;
      const float val = hacc[t].f[r] * zinv;
      const size_t idx = ((size_t)b * C_ + c) * M_ + m;
      hTf[idx] = val;
      hTb[idx] = f2b(val);
    }
}

// ------- per-layer GEMM yT = hT @ Wf^T (all operands K-contiguous in global) -------
__global__ __launch_bounds__(128) void filt_gemm_kernel(
    const u16* __restrict__ Wl, const u16* __restrict__ hTb,
    float* __restrict__ yT, float* __restrict__ stats)
{
  const int b = blockIdx.x, oblk = blockIdx.y;
  const int tid = threadIdx.x, wave = tid >> 5, lane = tid & 31;
  const int half8 = (lane >= 16) ? 8 : 0, col = lane & 15;
  const int h8 = half8, k16 = (lane >= 16) ? 16 : 0;

  CD acc[8];   // [cs(2)][ot(4)]
  #pragma unroll
  for (int t = 0; t < 8; ++t)
    #pragma unroll
    for (int r = 0; r < 8; ++r) acc[t].f[r] = 0.f;

  for (int ks = 0; ks < 16; ++ks) {       // K = M = 512
    AB a[2];
    #pragma unroll
    for (int cs = 0; cs < 2; ++cs) {
      const int c = wave * 32 + cs * 16 + col;
      load_afrag_g(a[cs], hTb + ((size_t)b * C_ + c) * M_ + ks * 32, h8);
    }
    #pragma unroll
    for (int ot = 0; ot < 4; ++ot) {
      AB bb;
      const int o = oblk * 64 + ot * 16 + col;
      load_bfrag_g(bb, Wl + (size_t)o * M_ + ks * 32, k16);
      acc[ot].v     = wmma_bf16(a[0].v, bb.v, acc[ot].v);
      acc[4 + ot].v = wmma_bf16(a[1].v, bb.v, acc[4 + ot].v);
    }
  }

  float s1[4] = {0, 0, 0, 0};
  float s2[4] = {0, 0, 0, 0};
  #pragma unroll
  for (int cs = 0; cs < 2; ++cs)
    #pragma unroll
    for (int ot = 0; ot < 4; ++ot)
      #pragma unroll
      for (int r = 0; r < 8; ++r) {
        const float v = acc[cs * 4 + ot].f[r];
        const int c = wave * 32 + cs * 16 + r + half8;
        yT[((size_t)b * C_ + c) * M_ + oblk * 64 + ot * 16 + col] = v;
        s1[ot] += v;
        s2[ot] += v * v;
      }
  #pragma unroll
  for (int ot = 0; ot < 4; ++ot) {
    float a1 = s1[ot] + __shfl_xor(s1[ot], 16);
    float a2 = s2[ot] + __shfl_xor(s2[ot], 16);
    if (lane < 16) {
      const int o = oblk * 64 + ot * 16 + lane;
      atomicAdd(&stats[o], a1);
      atomicAdd(&stats[M_ + o], a2);
    }
  }
}

// ---------------- BN + residual + ReLU (on transposed [B][C][M] layout) ----------------
__global__ __launch_bounds__(256) void filt_bn_kernel(
    const float* __restrict__ yT, const float* __restrict__ stats,
    const float* __restrict__ gamma, const float* __restrict__ beta,
    float* __restrict__ hTf, u16* __restrict__ hTb)
{
  const size_t i = (size_t)blockIdx.x * blockDim.x + threadIdx.x;
  if (i >= (size_t)B_ * C_ * M_) return;
  const int o = (int)(i % M_);
  const float inv = 1.f / (float)(B_ * C_);
  const float mu = stats[o] * inv;
  const float var = stats[M_ + o] * inv - mu * mu;
  const float yn = (yT[i] - mu) * rsqrtf(var + 1e-5f) * gamma[o] + beta[o];
  const float nh = fmaxf(yn + hTf[i], 0.f);
  hTf[i] = nh;
  hTb[i] = f2b(nh);
}

// ------- fused unpool: logits -> exp -> out = hT @ E, column-normalize -------
__global__ __launch_bounds__(128) void unpool_kernel(
    const u16* __restrict__ xsbT, const u16* __restrict__ Wub,
    const float* __restrict__ bu, const u16* __restrict__ hTb,
    float* __restrict__ out)
{
  const int b = blockIdx.x;
  const int n0 = blockIdx.y * NT2;
  const int tid = threadIdx.x, wave = tid >> 5, lane = tid & 31;
  const int half8 = (lane >= 16) ? 8 : 0, col = lane & 15;
  const int h8 = half8, k16 = (lane >= 16) ? 16 : 0;

  __shared__ u16 tile_nc[NT2][CP];   // [n][c] : K=c contiguous (logits B)
  __shared__ u16 et[NT2][NP];        // E^T chunk [n][m_local] : K=m contiguous
  __shared__ float csum[NT2];

  {
    const int rr = tid >> 1, hh = (tid & 1) * 64;
    const u16* g = xsbT + ((size_t)b * N_ + n0 + rr) * C_ + hh;
    const unsigned l = (unsigned)(size_t)&tile_nc[rr][hh];
    #pragma unroll
    for (int i = 0; i < 8; ++i) async_ld_b128(l + i * 16, g + i * 8);
    if (tid < NT2) csum[tid] = 0.f;
  }
  WAIT_ASYNC(0);
  __syncthreads();

  CD acc[8];   // [cs(2)][nt(4)]
  #pragma unroll
  for (int t = 0; t < 8; ++t)
    #pragma unroll
    for (int r = 0; r < 8; ++r) acc[t].f[r] = 0.f;

  for (int mc = 0; mc < M_; mc += 64) {
    // ul chunk: this wave's 16 m-rows x 64 n, K = C = 128
    CD lt[4];
    #pragma unroll
    for (int t = 0; t < 4; ++t)
      #pragma unroll
      for (int r = 0; r < 8; ++r) lt[t].f[r] = 0.f;

    #pragma unroll
    for (int kc = 0; kc < 4; ++kc) {
      AB a;
      load_afrag_g(a, Wub + (size_t)(mc + wave * 16 + col) * C_ + kc * 32, h8);
      #pragma unroll
      for (int t = 0; t < 4; ++t) {
        AB bb;
        load_bfrag_g(bb, &tile_nc[t * 16 + col][kc * 32], k16);
        lt[t].v = wmma_bf16(a.v, bb.v, lt[t].v);
      }
    }

    __syncthreads();   // previous chunk's et readers are done
    #pragma unroll
    for (int t = 0; t < 4; ++t) {
      float cp = 0.f;
      #pragma unroll
      for (int r = 0; r < 8; ++r) {
        const float e = __expf(lt[t].f[r] + bu[mc + wave * 16 + r + half8]);
        cp += e;
        et[t * 16 + col][wave * 16 + r + half8] = f2b(e);   // store transposed [n][m]
      }
      cp += __shfl_xor(cp, 16);
      if (lane < 16) atomicAdd(&csum[t * 16 + col], cp);    // ds_add_f32
    }
    __syncthreads();   // et chunk visible to all waves

    // acc(c x n) += hT(32c x 64m) @ E(64m x 64n)
    #pragma unroll
    for (int ks = 0; ks < 2; ++ks) {
      AB a[2];
      #pragma unroll
      for (int cs = 0; cs < 2; ++cs) {
        const int c = wave * 32 + cs * 16 + col;
        load_afrag_g(a[cs], hTb + ((size_t)b * C_ + c) * M_ + mc + ks * 32, h8);
      }
      #pragma unroll
      for (int t = 0; t < 4; ++t) {
        AB bb;
        load_bfrag_g(bb, &et[t * 16 + col][ks * 32], k16);
        acc[t].v     = wmma_bf16(a[0].v, bb.v, acc[t].v);
        acc[4 + t].v = wmma_bf16(a[1].v, bb.v, acc[4 + t].v);
      }
    }
  }
  __syncthreads();

  #pragma unroll
  for (int cs = 0; cs < 2; ++cs)
    #pragma unroll
    for (int t = 0; t < 4; ++t) {
      const float dinv = 1.f / csum[t * 16 + col];
      #pragma unroll
      for (int r = 0; r < 8; ++r) {
        const int c = wave * 32 + cs * 16 + r + half8;
        out[((size_t)b * C_ + c) * N_ + n0 + t * 16 + col] = acc[cs * 4 + t].f[r] * dinv;
      }
    }
}

// ---------------- host launcher ----------------
extern "C" void kernel_launch(void* const* d_in, const int* in_sizes, int n_in,
                              void* d_out, int out_size, void* d_ws, size_t ws_size,
                              hipStream_t stream)
{
  (void)in_sizes; (void)n_in; (void)out_size; (void)ws_size;
  const float* x  = (const float*)d_in[0];
  const float* Wp = (const float*)d_in[1];
  const float* Wf = (const float*)d_in[2];
  const float* ga = (const float*)d_in[3];
  const float* be = (const float*)d_in[4];
  const float* Wu = (const float*)d_in[5];
  const float* bu = (const float*)d_in[6];
  float* out = (float*)d_out;

  char* p = (char*)d_ws;
  auto carve = [&](size_t bytes) -> char* {
    char* r = p;
    p += (bytes + 255) & ~(size_t)255;
    return r;
  };
  u16*   xsb   = (u16*)  carve((size_t)B_ * C_ * N_ * sizeof(u16));
  u16*   xsbT  = (u16*)  carve((size_t)B_ * N_ * C_ * sizeof(u16));
  u16*   Wpb   = (u16*)  carve((size_t)M_ * C_ * sizeof(u16));
  u16*   Wub   = (u16*)  carve((size_t)M_ * C_ * sizeof(u16));
  u16*   Wfb   = (u16*)  carve((size_t)L_ * M_ * M_ * sizeof(u16));
  float* hTf   = (float*)carve((size_t)B_ * C_ * M_ * sizeof(float));
  u16*   hTb   = (u16*)  carve((size_t)B_ * C_ * M_ * sizeof(u16));
  float* yT    = (float*)carve((size_t)B_ * C_ * M_ * sizeof(float));
  float* stats = (float*)carve((size_t)L_ * 2 * M_ * sizeof(float));

  {
    const long long tot = (long long)M_ * C_ + (long long)L_ * M_ * M_ +
                          (long long)M_ * C_ + (long long)L_ * 2 * M_;
    const int blk = 256;
    const int grd = (int)((tot + blk - 1) / blk);
    prep_kernel<<<grd, blk, 0, stream>>>(Wp, Wf, Wu, Wpb, Wfb, Wub, stats);
  }
  convert_kernel<<<dim3(B_, C_ / 32, N_ / 32), dim3(32, 8), 0, stream>>>(x, xsb, xsbT);
  pool_kernel<<<dim3(B_, M_ / 64), 128, 0, stream>>>(xsb, xsbT, Wpb, hTf, hTb);
  for (int l = 0; l < L_; ++l) {
    filt_gemm_kernel<<<dim3(B_, M_ / 64), 128, 0, stream>>>(
        Wfb + (size_t)l * M_ * M_, hTb, yT, stats + (size_t)l * 2 * M_);
    filt_bn_kernel<<<(B_ * C_ * M_) / 256, 256, 0, stream>>>(
        yT, stats + (size_t)l * 2 * M_, ga + (size_t)l * M_, be + (size_t)l * M_,
        hTf, hTb);
  }
  unpool_kernel<<<dim3(B_, N_ / NT2), 128, 0, stream>>>(xsbT, Wub, bu, hTb, out);
}
